// ImageLRU_2388001816825
// MI455X (gfx1250) — matrix operand order repaired and checked by reading, not compiled, not run on hardware
//
#include <hip/hip_runtime.h>

typedef __attribute__((ext_vector_type(16))) _Float16 v16h;
typedef __attribute__((ext_vector_type(8)))  float    v8f;
typedef __attribute__((ext_vector_type(4)))  int      v4i;

// Address-space-qualified pointee types for the async global->LDS builtin:
// arg0 is v4i in AS(1) (global), arg1 is v4i in AS(3) (LDS).
typedef v4i __attribute__((address_space(1))) gv4i;
typedef v4i __attribute__((address_space(3))) lv4i;

#define L16   16
#define N64   64
#define NPIX  32768            // B*C*H*W = 2*4*64*64
#define WPB   2                // waves per block
#define PPW   8                // pixels per wave
#define NBLK  (NPIX / (WPB * PPW))   // 2048 blocks

// DS ops from one wave complete in order (ISA 7.3); this fence pins the
// compiler's ordering across LDS phase boundaries and drains dscnt.
#define LDS_FENCE()  asm volatile("s_wait_dscnt 0x0" ::: "memory")

#if defined(__gfx1250__) && __has_builtin(__builtin_amdgcn_global_load_async_to_lds_b128)
#define HAVE_ASYNC 1
#else
#define HAVE_ASYNC 0
#endif

#if HAVE_ASYNC
#define ASYNC_WAIT() asm volatile("s_wait_asynccnt 0x0" ::: "memory")
#else
#define ASYNC_WAIT() asm volatile("" ::: "memory")
#endif

__device__ __forceinline__ v8f wmma_f16f32(v16h a, v16h b, v8f c) {
  // (neg_a, A, neg_b, B, c_mod, C, reuse_a, reuse_b)
  return __builtin_amdgcn_wmma_f32_16x16x32_f16(false, a, false, b, (short)0, c,
                                                false, false);
}

// Stage one pixel's x (1024 f32) into the LDS stage plane.
// Async path: DMA writes LDS directly (no VGPR round trip), tracked by ASYNCcnt.
__device__ __forceinline__ void stage_x(const float* __restrict__ xin,
                                        float* stg, int lane) {
#if HAVE_ASYNC
#pragma unroll
  for (int j = 0; j < 8; ++j) {
    const int f = j * 128 + lane * 4;            // 512B contiguous per iter
    __builtin_amdgcn_global_load_async_to_lds_b128(
        (gv4i*)(xin + f), (lv4i*)(stg + f), 0, 0);
  }
#else
#pragma unroll
  for (int j = 0; j < 8; ++j) {
    const int f = j * 128 + lane * 4;
    *(float4*)(stg + f) = *(const float4*)(xin + f);
  }
#endif
}

__global__ __launch_bounds__(WPB * 32)
void imagelru_scan(const float* __restrict__ x,
                   const float* __restrict__ Lre, const float* __restrict__ Lim,
                   const float* __restrict__ Bre, const float* __restrict__ Bim,
                   const float* __restrict__ Cre, const float* __restrict__ Cim,
                   const float* __restrict__ Dm,
                   float* __restrict__ y)
{
  // Planes per wave: 0=s_re, 1=s_im, 2=t_re, 3=t_im, 4=stage (async target).
  // Fixed addresses -> compiler keeps immediate DS offsets and b128 vectorization.
  __shared__ __align__(16) float sbuf[WPB][5][N64 * L16];

  const int lane = threadIdx.x & 31;
  const int wid  = threadIdx.x >> 5;
  const int col  = lane & 15;   // N column for B/C/D layouts (= output k index)
  const int hf   = lane >> 4;   // lane half

  float* const s_re = &sbuf[wid][0][0];
  float* const s_im = &sbuf[wid][1][0];
  float* const t_re = &sbuf[wid][2][0];
  float* const t_im = &sbuf[wid][3][0];
  float* const stg  = &sbuf[wid][4][0];

  // ---- loop-invariant operands ----
  const float lam_r = Lre[col];
  const float lam_i = Lim[col];

  // B-operand (16-bit, 32x16): lane holds column N=col, element e -> l=hf*8+(e&7),
  // e<8 contracts with A's re part, e>=8 with A's im part.
  v16h fB_re, fB_im, fB_C, fB_D;
#pragma unroll
  for (int e = 0; e < 16; ++e) {
    const int mi = col * 16 + hf * 8 + (e & 7);   // matrix[k=col][l], row-major
    const bool ip = (e >= 8);
    fB_re[e] = (_Float16)(ip ? -Bim[mi] : Bre[mi]);   // re·Bre − im·Bim
    fB_im[e] = (_Float16)(ip ?  Bre[mi] : Bim[mi]);   // re·Bim + im·Bre
    fB_C [e] = (_Float16)(ip ? -Cim[mi] : Cre[mi]);   // Re(Cc·out)
    fB_D [e] = (_Float16)(ip ? 0.0f     : Dm [mi]);   // D·img (K=16 only)
  }

  const int gw = blockIdx.x * WPB + wid;
  const int nw = NBLK * WPB;

  // Prime the pipeline: async-prefetch first pixel into the stage plane.
  stage_x(x + (size_t)gw * (N64 * L16), stg, lane);

  for (int pix = gw; pix < NPIX; pix += nw) {
    float* yout = y + (size_t)pix * (N64 * L16);

    // Stage plane is complete; unpack it into the start buffer (re plane) and
    // zero the im plane. All immediate-offset b128 DS traffic.
    ASYNC_WAIT();
#pragma unroll
    for (int j = 0; j < 8; ++j) {
      const int f = j * 128 + lane * 4;
      const float4 v = *(const float4*)(stg + f);
      *(float4*)(s_re + f) = v;
      *(float4*)(s_im + f) = make_float4(0.f, 0.f, 0.f, 0.f);
    }
    // Copy reads must complete before the async engine may overwrite stage
    // (async ops are unordered w.r.t. DS ops).
    LDS_FENCE();

    // Launch next pixel's prefetch now; it overlaps the entire scan below.
    const int nxtp = pix + nw;                    // uniform per wave
    if (nxtp < NPIX) stage_x(x + (size_t)nxtp * (N64 * L16), stg, lane);

    // ---- D @ img, kept in registers (C-layout) through the scan ----
    v8f acc[4];
#pragma unroll
    for (int t = 0; t < 4; ++t) {
      const int abase = (t * 16 + col) * 16 + hf * 8;  // A row M=lane&15
      v16h a;
#pragma unroll
      for (int e = 0; e < 8; ++e) {
        a[e]     = (_Float16)s_re[abase + e];          // K = hf*8+e
        a[e + 8] = (_Float16)0.0f;                     // K = 16+hf*8+e (unused)
      }
      v8f z = {};
      acc[t] = wmma_f16f32(a, fB_D, z);
    }

    // ---- 6 scan levels, ping-pong s <-> t (even count: final lands in s) ----
#pragma unroll
    for (int d = 0; d < 6; ++d) {
      const int step = 1 << d;
      const float* cre = (d & 1) ? t_re : s_re;
      const float* cim = (d & 1) ? t_im : s_im;
      float*       nre = (d & 1) ? s_re : t_re;
      float*       nim = (d & 1) ? s_im : t_im;
#pragma unroll
      for (int t = 0; t < 4; ++t) {
        const int n0 = t * 16;
        if (n0 + 16 <= step) {                     // tile entirely frozen: copy
#pragma unroll
          for (int r = 0; r < 8; ++r) {
            const int idx = (n0 + r + hf * 8) * 16 + col;
            nre[idx] = cre[idx];
            nim[idx] = cim[idx];
          }
          continue;                                // uniform branch
        }
        // A = [right_re | right_im], rows of this tile
        const int abase = (n0 + col) * 16 + hf * 8;
        v16h a;
#pragma unroll
        for (int e = 0; e < 8; ++e) {
          a[e]     = (_Float16)cre[abase + e];
          a[e + 8] = (_Float16)cim[abase + e];
        }
        // C accumulator = Lambda ⊙ left (complex, f32)
        v8f c_r, c_i;
#pragma unroll
        for (int r = 0; r < 8; ++r) {
          int nl = n0 + r + hf * 8 - step;
          if (nl < 0) nl = 0;                      // masked at store time
          const float lr = cre[nl * 16 + col];
          const float li = cim[nl * 16 + col];
          c_r[r] = lam_r * lr - lam_i * li;
          c_i[r] = lam_r * li + lam_i * lr;
        }
        const v8f d_r = wmma_f16f32(a, fB_re, c_r);
        const v8f d_i = wmma_f16f32(a, fB_im, c_i);
#pragma unroll
        for (int r = 0; r < 8; ++r) {
          const int n   = n0 + r + hf * 8;
          const int idx = n * 16 + col;
          if (n >= step) { nre[idx] = d_r[r];  nim[idx] = d_i[r];  }
          else           { nre[idx] = cre[idx]; nim[idx] = cim[idx]; }
        }
      }
      LDS_FENCE();
    }

    // ---- y = Re(Cc @ out) + (D @ img already in acc); final state is in s ----
#pragma unroll
    for (int t = 0; t < 4; ++t) {
      const int abase = (t * 16 + col) * 16 + hf * 8;
      v16h a;
#pragma unroll
      for (int e = 0; e < 8; ++e) {
        a[e]     = (_Float16)s_re[abase + e];
        a[e + 8] = (_Float16)s_im[abase + e];
      }
      const v8f o = wmma_f16f32(a, fB_C, acc[t]);
#pragma unroll
      for (int r = 0; r < 8; ++r) {
        const int n = t * 16 + r + hf * 8;
        yout[n * 16 + col] = o[r];                 // 64B rows, coalesced
      }
    }
    LDS_FENCE();                                   // s/t reused next iteration
  }
}

extern "C" void kernel_launch(void* const* d_in, const int* in_sizes, int n_in,
                              void* d_out, int out_size, void* d_ws, size_t ws_size,
                              hipStream_t stream) {
  (void)in_sizes; (void)n_in; (void)out_size; (void)d_ws; (void)ws_size;
  const float* x   = (const float*)d_in[0];
  const float* Lre = (const float*)d_in[1];
  const float* Lim = (const float*)d_in[2];
  const float* Bre = (const float*)d_in[3];
  const float* Bim = (const float*)d_in[4];
  const float* Cre = (const float*)d_in[5];
  const float* Cim = (const float*)d_in[6];
  const float* Dm  = (const float*)d_in[7];
  float* y = (float*)d_out;

  imagelru_scan<<<dim3(NBLK), dim3(WPB * 32), 0, stream>>>(
      x, Lre, Lim, Bre, Bim, Cre, Cim, Dm, y);
}
